// Net_274877907721
// MI455X (gfx1250) — compile-verified, matrix operand order checked by reading, not run on hardware
//
#include <hip/hip_runtime.h>
#include <cstddef>

#define Bdim 64
#define Tdim 512
#define Idim 256
#define Hdim 1024
#define Odim 256
#define G4H  (4 * Hdim)

typedef _Float16 h16;
typedef __attribute__((ext_vector_type(16))) _Float16 v16h;
typedef __attribute__((ext_vector_type(8)))  _Float16 v8h;
typedef __attribute__((ext_vector_type(8)))  float    v8f;

union AFrag { v16h v; v8h h[2]; };

// Load a 16x32 f16 WMMA operand (A from row-major [16][K] slab, or B from
// row-major [N][K] weight rows == column-major 32x16 B). Per ISA 7.12.2:
// lane L<16 holds row L, K = kbase+0..7 and kbase+16..23;
// lane L>=16 holds row L-16, K = kbase+8..15 and kbase+24..31.
__device__ __forceinline__ v16h load_frag16x32(const h16* __restrict__ slab,
                                               int ld, int kbase, int lane)
{
    const int row = lane & 15;
    const int ko  = (lane >> 4) << 3;          // 0 or 8
    const h16* p  = slab + (size_t)row * ld + kbase + ko;
    AFrag f;
    f.h[0] = *(const v8h*)(p);                 // K = kbase + ko .. +7
    f.h[1] = *(const v8h*)(p + 16);            // K = kbase + 16 + ko .. +7
    return f.v;
}

__device__ __forceinline__ float sigf(float x)
{ return 1.0f / (1.0f + __expf(-x)); }
__device__ __forceinline__ float tanhf_fast(float x)
{ return 1.0f - 2.0f / (__expf(2.0f * x) + 1.0f); }

// ---------------- prep kernels ----------------
__global__ void k_f32_to_f16(const float* __restrict__ s, h16* __restrict__ d, int n)
{
    int i = blockIdx.x * blockDim.x + threadIdx.x;
    if (i < n) d[i] = (h16)s[i];
}

__global__ void k_bias_sum(const float* __restrict__ a, const float* __restrict__ b,
                           float* __restrict__ d, int n)
{
    int i = blockIdx.x * blockDim.x + threadIdx.x;
    if (i < n) d[i] = a[i] + b[i];
}

__global__ void k_zero_state(h16* __restrict__ h, float* __restrict__ c, int n)
{
    int i = blockIdx.x * blockDim.x + threadIdx.x;
    if (i < n) { h[i] = (h16)0.0f; c[i] = 0.0f; }
}

// ---------------- encoder LSTM cell (one time step) ----------------
// pre[m, g*H+n] = sum_k h_in[m,k]*Whh[g*H+n,k] + sum_k x[m,t,k]*Wih[g*H+n,k] + bias
// One wave per 16x16 (M x h-col) tile, 4 gate accumulators.
// 256 waves total: tile_m = wid & 3, tile_n = wid >> 2.
__global__ void k_enc_cell(const h16* __restrict__ h_in,    // [64][1024] f16
                           h16* __restrict__ h_out,         // [64][1024] f16
                           float* __restrict__ cbuf,        // [64][1024] f32 (in-place)
                           const h16* __restrict__ x_h,     // [B][T][I] f16
                           const h16* __restrict__ Whh,     // [4096][1024] f16
                           const h16* __restrict__ Wih,     // [4096][256] f16
                           const float* __restrict__ bias,  // [4096] = bih+bhh
                           int t)
{
    const int wid  = (blockIdx.x * blockDim.x + threadIdx.x) >> 5;
    const int lane = threadIdx.x & 31;
    const int m0 = (wid & 3) << 4;       // 0..48
    const int n0 = (wid >> 2) << 4;      // 0..1008

    v8f acc[4] = {};

    // recurrent part: K = H
    const h16* Aslab = h_in + (size_t)m0 * Hdim;
    for (int k = 0; k < Hdim; k += 32) {
        v16h a = load_frag16x32(Aslab, Hdim, k, lane);
#pragma unroll
        for (int g = 0; g < 4; ++g) {
            const h16* Bslab = Whh + (size_t)(g * Hdim + n0) * Hdim;
            v16h b = load_frag16x32(Bslab, Hdim, k, lane);
            acc[g] = __builtin_amdgcn_wmma_f32_16x16x32_f16(
                false, a, false, b, (short)0, acc[g], false, false);
        }
    }

    // input part: K = I; row m (=batch b) lives at x[(m*T + t)*I]
    const h16* Xslab = x_h + ((size_t)m0 * Tdim + t) * Idim;  // row stride T*I
    for (int k = 0; k < Idim; k += 32) {
        v16h a = load_frag16x32(Xslab, Tdim * Idim, k, lane);
#pragma unroll
        for (int g = 0; g < 4; ++g) {
            const h16* Bslab = Wih + (size_t)(g * Hdim + n0) * Idim;
            v16h b = load_frag16x32(Bslab, Idim, k, lane);
            acc[g] = __builtin_amdgcn_wmma_f32_16x16x32_f16(
                false, a, false, b, (short)0, acc[g], false, false);
        }
    }

    // gate math; D layout: vgpr r, lane l -> m = m0 + (l>>4)*8 + r, n = n0 + (l&15)
    const int n  = n0 + (lane & 15);
    const float bi = bias[n];
    const float bf = bias[Hdim + n];
    const float bg = bias[2 * Hdim + n];
    const float bo = bias[3 * Hdim + n];
    const int mr0 = m0 + ((lane >> 4) << 3);
#pragma unroll
    for (int r = 0; r < 8; ++r) {
        const size_t idx = (size_t)(mr0 + r) * Hdim + n;
        float gi = sigf(acc[0][r] + bi);
        float gf = sigf(acc[1][r] + bf);
        float gg = tanhf_fast(acc[2][r] + bg);
        float go = sigf(acc[3][r] + bo);
        float cn = gf * cbuf[idx] + gi * gg;
        cbuf[idx]  = cn;
        h_out[idx] = (h16)(go * tanhf_fast(cn));
    }
}

// ---------------- decoder: batched single-step cell over all (b,t) ----------------
// pre[m, g*H+n] = sum_k x[m,k]*Wih[g*H+n,k] + dbase[b, g*H+n],  m = b*T + t
// tiles_m = 2048, tiles_n = 64; tile_n = wid / 2048 so 4 block-mate waves share B.
__global__ void k_dec_cell(const h16* __restrict__ x_h,     // [32768][256] f16
                           const h16* __restrict__ Wih,     // [4096][256] f16
                           const float* __restrict__ dbase, // [64][4096] f32
                           const float* __restrict__ cfin,  // [64][1024] f32
                           h16* __restrict__ hdec)          // [32768][1024] f16
{
    const int wid  = (blockIdx.x * blockDim.x + threadIdx.x) >> 5;
    const int lane = threadIdx.x & 31;
    const int tile_m = wid & 2047;
    const int tile_n = wid >> 11;
    const int m0 = tile_m << 4;
    const int n0 = tile_n << 4;

    v8f acc[4] = {};
    const h16* Aslab = x_h + (size_t)m0 * Idim;
    for (int k = 0; k < Idim; k += 32) {
        v16h a = load_frag16x32(Aslab, Idim, k, lane);
#pragma unroll
        for (int g = 0; g < 4; ++g) {
            const h16* Bslab = Wih + (size_t)(g * Hdim + n0) * Idim;
            v16h b = load_frag16x32(Bslab, Idim, k, lane);
            acc[g] = __builtin_amdgcn_wmma_f32_16x16x32_f16(
                false, a, false, b, (short)0, acc[g], false, false);
        }
    }

    const int n = n0 + (lane & 15);
    const int bidx = m0 / Tdim;                 // 16-row tile never crosses a batch
    const float di = dbase[(size_t)bidx * G4H + n];
    const float df = dbase[(size_t)bidx * G4H + Hdim + n];
    const float dg = dbase[(size_t)bidx * G4H + 2 * Hdim + n];
    const float dz = dbase[(size_t)bidx * G4H + 3 * Hdim + n];
    const float co = cfin[(size_t)bidx * Hdim + n];
    const int mr0 = m0 + ((lane >> 4) << 3);
#pragma unroll
    for (int r = 0; r < 8; ++r) {
        float gi = sigf(acc[0][r] + di);
        float gf = sigf(acc[1][r] + df);
        float gg = tanhf_fast(acc[2][r] + dg);
        float go = sigf(acc[3][r] + dz);
        float cn = gf * co + gi * gg;
        hdec[(size_t)(mr0 + r) * Hdim + n] = (h16)(go * tanhf_fast(cn));
    }
}

// ---------------- generic WMMA GEMM + bias: C[M,N] = A[M,K] * B[N,K]^T + bias[N] ----------------
__global__ void k_gemm_bias(const h16* __restrict__ A, int lda,
                            const h16* __restrict__ Bw, int ldb,
                            const float* __restrict__ bias,
                            float* __restrict__ C, int ldc,
                            int tiles_m, int K)
{
    const int wid  = (blockIdx.x * blockDim.x + threadIdx.x) >> 5;
    const int lane = threadIdx.x & 31;
    const int m0 = (wid % tiles_m) << 4;
    const int n0 = (wid / tiles_m) << 4;

    v8f acc = {};
    const h16* Aslab = A  + (size_t)m0 * lda;
    const h16* Bslab = Bw + (size_t)n0 * ldb;
    for (int k = 0; k < K; k += 32) {
        v16h a = load_frag16x32(Aslab, lda, k, lane);
        v16h b = load_frag16x32(Bslab, ldb, k, lane);
        acc = __builtin_amdgcn_wmma_f32_16x16x32_f16(
            false, a, false, b, (short)0, acc, false, false);
    }
    const int n = n0 + (lane & 15);
    const float bv = bias ? bias[n] : 0.0f;
    const int mr0 = m0 + ((lane >> 4) << 3);
#pragma unroll
    for (int r = 0; r < 8; ++r)
        C[(size_t)(mr0 + r) * ldc + n] = acc[r] + bv;
}

// ---------------- host ----------------
extern "C" void kernel_launch(void* const* d_in, const int* in_sizes, int n_in,
                              void* d_out, int out_size, void* d_ws, size_t ws_size,
                              hipStream_t stream)
{
    const float* x        = (const float*)d_in[0];
    const float* enc_Wih  = (const float*)d_in[1];
    const float* enc_Whh  = (const float*)d_in[2];
    const float* enc_bih  = (const float*)d_in[3];
    const float* enc_bhh  = (const float*)d_in[4];
    const float* dec_Wih  = (const float*)d_in[5];
    const float* dec_Whh  = (const float*)d_in[6];
    const float* dec_bih  = (const float*)d_in[7];
    const float* dec_bhh  = (const float*)d_in[8];
    const float* fc_W     = (const float*)d_in[9];
    const float* fc_b     = (const float*)d_in[10];
    float* out            = (float*)d_out;

    char* ws = (char*)d_ws;
    size_t off = 0;
    auto carve = [&](size_t bytes) -> void* {
        void* p = ws + off;
        off = (off + bytes + 255) & ~(size_t)255;
        return p;
    };

    const size_t nX    = (size_t)Bdim * Tdim * Idim;   // 8388608
    const size_t nWihE = (size_t)G4H * Idim;           // 1048576
    const size_t nWhh  = (size_t)G4H * Hdim;           // 4194304
    const size_t nFcW  = (size_t)Odim * Hdim;          // 262144
    const size_t nHC   = (size_t)Bdim * Hdim;          // 65536
    const size_t nHdec = (size_t)Bdim * Tdim * Hdim;   // 33554432

    h16*   x_h    = (h16*)  carve(nX    * sizeof(h16));
    h16*   eWih_h = (h16*)  carve(nWihE * sizeof(h16));
    h16*   eWhh_h = (h16*)  carve(nWhh  * sizeof(h16));
    h16*   dWih_h = (h16*)  carve(nWihE * sizeof(h16));
    h16*   dWhh_h = (h16*)  carve(nWhh  * sizeof(h16));
    h16*   fcW_h  = (h16*)  carve(nFcW  * sizeof(h16));
    float* bias_e = (float*)carve(G4H   * sizeof(float));
    float* bias_d = (float*)carve(G4H   * sizeof(float));
    h16*   h0     = (h16*)  carve(nHC   * sizeof(h16));
    h16*   h1     = (h16*)  carve(nHC   * sizeof(h16));
    float* cbuf   = (float*)carve(nHC   * sizeof(float));
    float* dbase  = (float*)carve((size_t)Bdim * G4H * sizeof(float));
    h16*   hdec   = (h16*)  carve(nHdec * sizeof(h16));
    (void)ws_size; (void)in_sizes; (void)n_in; (void)out_size;

    auto cvt = [&](const float* s, h16* d, size_t n) {
        k_f32_to_f16<<<(unsigned)((n + 255) / 256), 256, 0, stream>>>(s, d, (int)n);
    };
    cvt(x,       x_h,    nX);
    cvt(enc_Wih, eWih_h, nWihE);
    cvt(enc_Whh, eWhh_h, nWhh);
    cvt(dec_Wih, dWih_h, nWihE);
    cvt(dec_Whh, dWhh_h, nWhh);
    cvt(fc_W,    fcW_h,  nFcW);
    k_bias_sum<<<(G4H + 255) / 256, 256, 0, stream>>>(enc_bih, enc_bhh, bias_e, G4H);
    k_bias_sum<<<(G4H + 255) / 256, 256, 0, stream>>>(dec_bih, dec_bhh, bias_d, G4H);
    k_zero_state<<<(unsigned)((nHC + 255) / 256), 256, 0, stream>>>(h0, cbuf, (int)nHC);

    // encoder recurrence: 512 dependent steps, ping-pong h (cross-block RAW safety)
    for (int t = 0; t < Tdim; ++t) {
        const h16* hin = (t & 1) ? h1 : h0;
        h16*      hout = (t & 1) ? h0 : h1;
        // 256 waves = 64 blocks x 4 waves (wave32)
        k_enc_cell<<<64, 128, 0, stream>>>(hin, hout, cbuf, x_h,
                                           eWhh_h, eWih_h, bias_e, t);
    }
    // after t=511 (odd), final h is in h0; c final is cbuf (in place)

    // dec_base[b, :] = h_final @ dec_Whh^T + bih + bhh   (1024 waves)
    k_gemm_bias<<<256, 128, 0, stream>>>(h0, Hdim, dWhh_h, Hdim, bias_d,
                                         dbase, G4H, /*tiles_m=*/4, /*K=*/Hdim);

    // decoder batched cell over all (b,t): 131072 waves = 32768 blocks x 4
    k_dec_cell<<<32768, 128, 0, stream>>>(x_h, dWih_h, dbase, cbuf, hdec);

    // pred = hdec @ fc_W^T + fc_b : M=32768, N=256 -> 32768 waves
    k_gemm_bias<<<8192, 128, 0, stream>>>(hdec, Hdim, fcW_h, Hdim, fc_b,
                                          out, Odim, /*tiles_m=*/2048, /*K=*/Hdim);
}
